// MultiheadAttention_with_pos_embed_66331474920203
// MI455X (gfx1250) — matrix operand order repaired and checked by reading, not compiled, not run on hardware
//
#include <hip/hip_runtime.h>

typedef _Float16 v16h __attribute__((ext_vector_type(16)));
typedef _Float16 v8h  __attribute__((ext_vector_type(8)));
typedef _Float16 v4h  __attribute__((ext_vector_type(4)));
typedef float    v8f  __attribute__((ext_vector_type(8)));
typedef unsigned int u32x4 __attribute__((ext_vector_type(4)));
typedef int          i32x8 __attribute__((ext_vector_type(8)));

#define BSZ   2
#define SEQ   2048
#define CDIM  1024
#define NH    16
#define HD    64
#define MTOT  (BSZ * SEQ)   // 4096

static __device__ __forceinline__ v16h cat8(v8h a, v8h b) {
    return __builtin_shufflevector(a, b, 0,1,2,3,4,5,6,7,8,9,10,11,12,13,14,15);
}

// Wave-relative LDS byte address: generic pointers to LDS carry the LDS
// offset in their low 32 bits.
static __device__ __forceinline__ unsigned lds_addr32(const void* p) {
    return (unsigned)(size_t)p;
}

// Async DMA: global memory -> LDS, 16 bytes, tracked with ASYNCcnt.
static __device__ __forceinline__ void async_load_b128(unsigned lds, const void* gaddr) {
    asm volatile("global_load_async_to_lds_b128 %0, %1, off"
                 :: "v"(lds), "v"(gaddr) : "memory");
}
static __device__ __forceinline__ void wait_asynccnt0() {
    asm volatile("s_wait_asynccnt 0x0" ::: "memory");
}

// Tensor Data Mover: wave-issued 2D tile DMA global->LDS (ISA 08 sect 7/8),
// 2-group descriptor form (VADDR2/VADDR3 = NULL => up-to-2D tensors).
// Tracked with TENSORcnt.
static __device__ __forceinline__ void tdm_load_2d(u32x4 g0, i32x8 g1) {
    asm volatile("tensor_load_to_lds %0, %1" :: "s"(g0), "s"(g1) : "memory");
}

// D# group 0: count=1 valid descriptor, LDS dest address, 57-bit global
// address of tile start, type=2 ("image") in bits [127:126].
static __device__ __forceinline__ u32x4 tdm_group0(unsigned lds, const void* gaddr) {
    unsigned long long ga = (unsigned long long)gaddr;
    u32x4 g0;
    g0[0] = 1u;                                        // count=1
    g0[1] = lds;                                       // lds_addr
    g0[2] = (unsigned)(ga & 0xFFFFFFFFu);              // global_addr[31:0]
    g0[3] = (unsigned)((ga >> 32) & 0x1FFFFFFu)        // global_addr[56:32]
          | (2u << 30);                                // type = 2
    return g0;
}

// ---------------------------------------------------------------------------
// Kernel 1: fused projection GEMM (x @ W^T + b) + RoPE + f16 head-major pack.
//   mode 0: Q -> Qh[B,H,N,HD]   (with RoPE)
//   mode 1: K -> Kh[B,H,N,HD]   (with RoPE)
//   mode 2: V -> Vt[B,H,HD,N]   (transposed so PV B-fragments are contiguous)
// Block tile 128(M) x 64(N), 8 waves in a 4x2 grid, each wave 32x32 via
// 2x2 WMMA 16x16x32_f16 tiles. fp32 inputs converted to f16 while staging
// into padded LDS (row stride 40 halves => all fragment loads 16B aligned).
// ---------------------------------------------------------------------------
__global__ __launch_bounds__(256)
void proj_rope_kernel(const float* __restrict__ X, const float* __restrict__ W,
                      const float* __restrict__ bias,
                      const float* __restrict__ cosT, const float* __restrict__ sinT,
                      _Float16* __restrict__ out, int mode)
{
    __shared__ _Float16 Xs[128 * 40];
    __shared__ _Float16 Ws[64 * 40];

    const int tid  = threadIdx.x;
    const int lane = tid & 31;
    const int w    = tid >> 5;
    const int wm   = w >> 1;       // 0..3
    const int wn   = w & 1;        // 0..1
    const int M0   = blockIdx.x * 128;
    const int N0   = blockIdx.y * 64;
    const int hi   = lane >> 4;    // 0/1: which half of the wave
    const int lr   = lane & 15;

    v8f acc[2][2] = {};

    for (int k0 = 0; k0 < CDIM; k0 += 32) {
        __syncthreads();
        // stage X tile: 128x32 fp32 -> f16 LDS
        #pragma unroll
        for (int i = 0; i < 4; ++i) {
            int idx = tid + i * 256;
            int r = idx >> 3, c4 = (idx & 7) << 2;
            float4 f = *(const float4*)(X + (size_t)(M0 + r) * CDIM + k0 + c4);
            v4h h; h[0] = (_Float16)f.x; h[1] = (_Float16)f.y;
                   h[2] = (_Float16)f.z; h[3] = (_Float16)f.w;
            *(v4h*)(Xs + r * 40 + c4) = h;
        }
        // stage W tile: 64x32 fp32 -> f16 LDS (W is [N,K] row-major: K contiguous)
        #pragma unroll
        for (int i = 0; i < 2; ++i) {
            int idx = tid + i * 256;
            int r = idx >> 3, c4 = (idx & 7) << 2;
            float4 f = *(const float4*)(W + (size_t)(N0 + r) * CDIM + k0 + c4);
            v4h h; h[0] = (_Float16)f.x; h[1] = (_Float16)f.y;
                   h[2] = (_Float16)f.z; h[3] = (_Float16)f.w;
            *(v4h*)(Ws + r * 40 + c4) = h;
        }
        __syncthreads();

        // A fragments: lane holds row lr, chunks at k = hi*8 and 16+hi*8
        v16h a[2];
        #pragma unroll
        for (int mi = 0; mi < 2; ++mi) {
            const _Float16* p = Xs + (wm * 32 + mi * 16 + lr) * 40;
            a[mi] = cat8(*(const v8h*)(p + hi * 8), *(const v8h*)(p + 16 + hi * 8));
        }
        // B fragments: lane holds W-row lr of subtile, chunks at k = hi*8 and 16+hi*8
        v16h bf[2];
        #pragma unroll
        for (int ni = 0; ni < 2; ++ni) {
            const _Float16* p = Ws + (wn * 32 + ni * 16 + lr) * 40;
            bf[ni] = cat8(*(const v8h*)(p + hi * 8), *(const v8h*)(p + 16 + hi * 8));
        }
        #pragma unroll
        for (int mi = 0; mi < 2; ++mi)
            #pragma unroll
            for (int ni = 0; ni < 2; ++ni)
                acc[mi][ni] = __builtin_amdgcn_wmma_f32_16x16x32_f16(
                    false, a[mi], false, bf[ni], (short)0, acc[mi][ni], false, false);
    }

    // Epilogue: bias + RoPE (modes 0/1) + pack f16 into head-major workspace.
    #pragma unroll
    for (int ni = 0; ni < 2; ++ni) {
        const int n_g = N0 + wn * 32 + ni * 16 + lr;  // output channel
        const float bv = bias[n_g];
        const int hh = n_g >> 6;        // head
        const int j  = n_g & 63;        // channel within head
        const int pp = j >> 1;          // rope pair index
        #pragma unroll
        for (int mi = 0; mi < 2; ++mi) {
            #pragma unroll
            for (int r = 0; r < 8; ++r) {
                const int m = M0 + wm * 32 + mi * 16 + r + hi * 8;  // flat token
                const int bb = m >> 11;
                const int t  = m & (SEQ - 1);
                float val = acc[mi][ni][r] + bv;
                if (mode < 2) {
                    const float c = cosT[t * (HD / 2) + pp];
                    const float s = sinT[t * (HD / 2) + pp];
                    const float partner = __shfl_xor(val, 1);  // even<->odd channel
                    val = (n_g & 1) ? (partner * s + val * c)
                                    : (val * c - partner * s);
                    out[(((size_t)(bb * NH + hh)) * SEQ + t) * HD + j] = (_Float16)val;
                } else {
                    out[(((size_t)(bb * NH + hh)) * HD + j) * SEQ + t] = (_Float16)val;
                }
            }
        }
    }
}

// ---------------------------------------------------------------------------
// Kernel 2: flash attention. 8 waves/block; wave owns 16 query rows of one
// head. Stream 32-key tiles:
//   K tile (32x64 f16, padded LDS rows): one TDM descriptor issued by wave 0
//     (tensor_load_to_lds, TENSORcnt), hardware applies the 16B row padding.
//   V tile (64x32 f16): per-thread global_load_async_to_lds_b128 (ASYNCcnt).
// Then 4 WMMAs for S = Q K^T, online softmax with half-wave shuffle
// reductions, P via per-wave LDS round-trip, 4 WMMAs for O += P V.
// ---------------------------------------------------------------------------
__global__ __launch_bounds__(256)
void attn_kernel(const _Float16* __restrict__ Qh, const _Float16* __restrict__ Kh,
                 const _Float16* __restrict__ Vt, float* __restrict__ out)
{
    __shared__ _Float16 Ks[32 * 72];       // 32 keys x hd64, stride 72 (128B+16B pad)
    __shared__ _Float16 Vs[64 * 40];       // hd64 x 32 keys, stride 40
    __shared__ _Float16 Pl[8 * 16 * 40];   // per-wave P scratch, stride 40

    const int tid  = threadIdx.x;
    const int lane = tid & 31;
    const int w    = tid >> 5;
    const int hi   = lane >> 4;
    const int lr   = lane & 15;

    const int bh = blockIdx.y;             // 0..31 (b*16+h)
    const int b  = bh >> 4, h = bh & 15;
    const int qt0 = blockIdx.x * 128 + w * 16;

    // Q fragments for the wave's 16 rows, loaded once (K-contiguous in hd).
    const _Float16* Qbase = Qh + (((size_t)bh) * SEQ + qt0 + lr) * HD;
    v16h qf[2];
    #pragma unroll
    for (int kd = 0; kd < 2; ++kd)
        qf[kd] = cat8(*(const v8h*)(Qbase + kd * 32 + hi * 8),
                      *(const v8h*)(Qbase + kd * 32 + 16 + hi * 8));

    v8f o[4] = {};
    float mI[8], lI[8];
    #pragma unroll
    for (int r = 0; r < 8; ++r) { mI[r] = -__builtin_inff(); lI[r] = 0.f; }

    const _Float16* Kp = Kh + ((size_t)bh) * SEQ * HD;
    const _Float16* Vp = Vt + ((size_t)bh) * HD * SEQ;
    _Float16* Pw = Pl + w * 16 * 40;

    // TDM D# group 1 for the K tile (loop-invariant):
    //   data_size=2B, pad_enable, pad every 128B (pad_interval=4 -> 16x8B)
    //   by 16B (pad_amount=3 -> 4 DWORDs); tensor = 2048x64 halves,
    //   tile = 32 rows x 64 halves, row stride 64 elements.
    i32x8 g1K;
    g1K[0] = (1 << 16) | (1 << 20) | (4 << 22) | (3 << 25);
    g1K[1] = (int)(64u << 16);            // tensor_dim0[15:0] = 64
    g1K[2] = (int)(2048u << 16);          // dim0[31:16]=0 | tensor_dim1[15:0]=2048
    g1K[3] = (int)(64u << 16);            // dim1[31:16]=0 | tile_dim0 = 64
    g1K[4] = 32;                          // tile_dim1 = 32, tile_dim2 = 0
    g1K[5] = 64;                          // tensor_dim0_stride[31:0] = 64
    g1K[6] = 0;
    g1K[7] = 0;
    const unsigned ldsKbase = lds_addr32(Ks);

    // Per-thread staging coordinates for the V tile (one b128 async DMA each).
    const int vd = tid >> 2, vc = tid & 3;          // V: 64 rows x 4 chunks
    const unsigned ldsV = lds_addr32(Vs + vd * 40 + vc * 8);

    for (int n0 = 0; n0 < SEQ; n0 += 32) {
        __syncthreads();   // previous tile fully consumed by all waves
        if (w == 0)        // one TDM descriptor moves the whole K tile
            tdm_load_2d(tdm_group0(ldsKbase, Kp + (size_t)n0 * HD), g1K);
        // V tile via per-thread async DMA (no VGPR round-trip).
        async_load_b128(ldsV, Vp + (size_t)vd * SEQ + n0 + vc * 8);
        if (n0 + 32 < SEQ)
            __builtin_prefetch(Kp + (size_t)(n0 + 32) * HD + tid * 8, 0, 0);
        wait_asynccnt0();                       // this wave's async DMAs landed
        if (w == 0)
            __builtin_amdgcn_s_wait_tensorcnt(0);  // wave 0's TDM landed
        __syncthreads();   // publish K and V tiles to all waves

        // S = Q K^T for two 16-key subtiles
        v8f s[2] = {};
        #pragma unroll
        for (int nt = 0; nt < 2; ++nt) {
            #pragma unroll
            for (int kd = 0; kd < 2; ++kd) {
                const _Float16* p = Ks + (nt * 16 + lr) * 72 + kd * 32;
                v16h kf = cat8(*(const v8h*)(p + hi * 8), *(const v8h*)(p + 16 + hi * 8));
                s[nt] = __builtin_amdgcn_wmma_f32_16x16x32_f16(
                    false, qf[kd], false, kf, (short)0, s[nt], false, false);
            }
        }

        // Online softmax (scale = hd^-0.5 = 0.125). Row stats reduce across
        // the 16 lanes of each wave-half; slot r maps identically in S and O.
        #pragma unroll
        for (int r = 0; r < 8; ++r) {
            float v0 = s[0][r] * 0.125f, v1 = s[1][r] * 0.125f;
            float rmax = fmaxf(v0, v1);
            #pragma unroll
            for (int msk = 1; msk < 16; msk <<= 1)
                rmax = fmaxf(rmax, __shfl_xor(rmax, msk));
            const float mnew  = fmaxf(mI[r], rmax);
            const float alpha = __expf(mI[r] - mnew);
            const float p0 = __expf(v0 - mnew), p1 = __expf(v1 - mnew);
            float rsum = p0 + p1;
            #pragma unroll
            for (int msk = 1; msk < 16; msk <<= 1)
                rsum += __shfl_xor(rsum, msk);
            lI[r] = lI[r] * alpha + rsum;
            mI[r] = mnew;
            #pragma unroll
            for (int sub = 0; sub < 4; ++sub) o[sub][r] *= alpha;
            const int row = r + hi * 8;     // C-layout -> LDS (row-major P)
            Pw[row * 40 + lr]      = (_Float16)p0;
            Pw[row * 40 + 16 + lr] = (_Float16)p1;
        }
        asm volatile("s_wait_dscnt 0" ::: "memory");

        // Reload P in A-fragment layout (per-wave region: no barrier needed;
        // same-wave DS ops are serviced in order).
        v16h pf;
        {
            const _Float16* p = Pw + lr * 40;
            pf = cat8(*(const v8h*)(p + hi * 8), *(const v8h*)(p + 16 + hi * 8));
        }
        // O += P V over four 16-wide d subtiles (Vs rows are d, cols are keys)
        #pragma unroll
        for (int sub = 0; sub < 4; ++sub) {
            const _Float16* p = Vs + (sub * 16 + lr) * 40;
            v16h vf = cat8(*(const v8h*)(p + hi * 8), *(const v8h*)(p + 16 + hi * 8));
            o[sub] = __builtin_amdgcn_wmma_f32_16x16x32_f16(
                false, pf, false, vf, (short)0, o[sub], false, false);
        }
    }

    // Normalize and write out [B, N, C] fp32.
    #pragma unroll
    for (int r = 0; r < 8; ++r) {
        const float inv = 1.f / lI[r];
        const int t = qt0 + r + hi * 8;
        const size_t base = ((size_t)b * SEQ + t) * CDIM + h * HD;
        #pragma unroll
        for (int sub = 0; sub < 4; ++sub)
            out[base + sub * 16 + lr] = o[sub][r] * inv;
    }
}

// ---------------------------------------------------------------------------
extern "C" void kernel_launch(void* const* d_in, const int* in_sizes, int n_in,
                              void* d_out, int out_size, void* d_ws, size_t ws_size,
                              hipStream_t stream)
{
    const float* q    = (const float*)d_in[0];
    const float* k    = (const float*)d_in[1];
    const float* v    = (const float*)d_in[2];
    const float* qcos = (const float*)d_in[3];
    const float* qsin = (const float*)d_in[4];
    const float* kcos = (const float*)d_in[5];
    const float* ksin = (const float*)d_in[6];
    const float* Wq   = (const float*)d_in[7];
    const float* bq   = (const float*)d_in[8];
    const float* Wk   = (const float*)d_in[9];
    const float* bk   = (const float*)d_in[10];
    const float* Wv   = (const float*)d_in[11];
    const float* bv   = (const float*)d_in[12];
    float* out = (float*)d_out;

    const size_t perTensor = (size_t)BSZ * NH * SEQ * HD;   // 4M halves = 8 MB
    _Float16* Qh = (_Float16*)d_ws;
    _Float16* Kh = Qh + perTensor;
    _Float16* Vt = Kh + perTensor;

    dim3 g1(MTOT / 128, CDIM / 64);
    proj_rope_kernel<<<g1, 256, 0, stream>>>(q, Wq, bq, qcos, qsin, Qh, 0);
    proj_rope_kernel<<<g1, 256, 0, stream>>>(k, Wk, bk, kcos, ksin, Kh, 1);
    proj_rope_kernel<<<g1, 256, 0, stream>>>(v, Wv, bv, qcos, qsin, Vt, 2);

    dim3 g2(SEQ / 128, BSZ * NH);
    attn_kernel<<<g2, 256, 0, stream>>>(Qh, Kh, Vt, out);
}
